// MemPool_20933670600813
// MI455X (gfx1250) — compile-verified, matrix operand order checked by reading, not run on hardware
//
#include <hip/hip_runtime.h>
#include <hip/hip_bf16.h>

typedef __attribute__((ext_vector_type(16))) _Float16 v16h;
typedef __attribute__((ext_vector_type(8)))  float    v8f;

#define B_   8
#define N_   1024
#define F_   128
#define H_   4
#define K_   32
#define PTS  32   // points per block in the distance kernel

__device__ __forceinline__ float wave_sum32(float v) {
#pragma unroll
  for (int m = 16; m >= 1; m >>= 1) v += __shfl_xor(v, m, 32);
  return v;
}
__device__ __forceinline__ float wave_max32(float v) {
#pragma unroll
  for (int m = 16; m >= 1; m >>= 1) v = fmaxf(v, __shfl_xor(v, m, 32));
  return v;
}

// -------------------------------------------------------------------------
// Kernel 1: L1 distance -> student-t -> per-head normalize -> conv-combine
//           -> softmax  ==> S[b,n,k]
// grid = B_*(N_/PTS), block = 128 threads (wave == head, lane == k)
// thread tid owns key row hk = tid (kept in 128 VGPRs)
// -------------------------------------------------------------------------
__global__ __launch_bounds__(128)
void mempool_s_kernel(const float* __restrict__ x,
                      const float* __restrict__ keys,
                      const float* __restrict__ conv_w,
                      float* __restrict__ S) {
  __shared__ __align__(16) float xs[PTS][F_];
  __shared__ float cbuf[H_][K_];

  const int tid  = threadIdx.x;
  const int lane = tid & 31;
  const int head = tid >> 5;
  const int b    = blockIdx.x / (N_ / PTS);
  const int n0   = (blockIdx.x % (N_ / PTS)) * PTS;

  // stage x tile (PTS x F_) into LDS, float4-vectorized, coalesced
  {
    const float4* src = (const float4*)(x + ((size_t)b * N_ + n0) * F_);
    float4* dst = (float4*)&xs[0][0];
#pragma unroll
    for (int i = 0; i < (PTS * F_ / 4) / 128; ++i)
      dst[tid + i * 128] = src[tid + i * 128];
  }

  // this thread's key row in registers (fully unrolled -> stays in VGPRs)
  float kreg[F_];
  {
    const float4* kp = (const float4*)(keys + (size_t)tid * F_);
#pragma unroll
    for (int i = 0; i < F_ / 4; ++i) {
      float4 v = kp[i];
      kreg[4*i+0] = v.x; kreg[4*i+1] = v.y;
      kreg[4*i+2] = v.z; kreg[4*i+3] = v.w;
    }
  }
  const float cw = conv_w[head];
  __syncthreads();

  for (int p = 0; p < PTS; ++p) {
    // L1 distance: ds_load_b128 broadcast + v_sub/v_add|..| per element
    const float4* xp = (const float4*)&xs[p][0];
    float d = 0.f;
#pragma unroll
    for (int i = 0; i < F_ / 4; ++i) {
      float4 v = xp[i];
      d += fabsf(kreg[4*i+0] - v.x);
      d += fabsf(kreg[4*i+1] - v.y);
      d += fabsf(kreg[4*i+2] - v.z);
      d += fabsf(kreg[4*i+3] - v.w);
    }
    float w  = 1.0f / (1.0f + d * d);   // (1 + d^2/tau)^(-(tau+1)/2), tau=1
    float hs = wave_sum32(w);           // sum over K within this head (wave)
    cbuf[head][lane] = cw * (w / hs);   // head-normalized, conv-weighted
    __syncthreads();
    if (head == 0) {
      float c = cbuf[0][lane] + cbuf[1][lane] + cbuf[2][lane] + cbuf[3][lane];
      float m = wave_max32(c);
      float e = expf(c - m);
      float s = wave_sum32(e);
      S[((size_t)b * N_ + (n0 + p)) * K_ + lane] = e / s;
    }
    __syncthreads();
  }
}

// -------------------------------------------------------------------------
// Kernel 2 (fused GEMMs, f16 WMMA / f32 accum):
//   phase 1: pooled[32x128] = S^T[32x1024] * X[1024x128]   (per batch)
//   phase 2: out = LeakyReLU(pooled[32x128] @ lin_w^T[128x128])
// grid = B_, block = 256 (8 waves); wave w owns output columns w*16..w*16+15
// -------------------------------------------------------------------------
__device__ __forceinline__ int kmap16(int j, int half) {
  // ISA 16-bit A/B layout: element j of lane (half,col) holds
  // K = (j>=8)*16 + half*8 + (j&7)
  return ((j >> 3) << 4) + half * 8 + (j & 7);
}

__global__ __launch_bounds__(256)
void mempool_out_kernel(const float* __restrict__ x,
                        const float* __restrict__ lin_w,
                        const float* __restrict__ S,
                        float* __restrict__ out) {
  __shared__ __align__(16) float pLDS[K_][F_];   // pooled tile, 32x128 f32

  const int tid  = threadIdx.x;
  const int lane = tid & 31;
  const int wv   = tid >> 5;
  const int half = lane >> 4;
  const int col  = lane & 15;
  const int b    = blockIdx.x;
  const int fcol = wv * 16 + col;

  const float* Sb = S + (size_t)b * N_ * K_;
  const float* xb = x + (size_t)b * N_ * F_;

  // ---- phase 1: pooled = S^T * X, K-dim = n (1024), 32 WMMA steps x 2 tiles
  v8f acc0 = {}; v8f acc1 = {};
  for (int kb = 0; kb < N_ / 32; ++kb) {
    const int nbase = kb * 32;
    v16h a0, a1, bf;
#pragma unroll
    for (int j = 0; j < 16; ++j) {
      const int n = nbase + kmap16(j, half);
      bf[j] = (_Float16)xb[(size_t)n * F_ + fcol];        // B[n][fcol]
      a0[j] = (_Float16)Sb[(size_t)n * K_ + col];         // A=S^T rows 0..15
      a1[j] = (_Float16)Sb[(size_t)n * K_ + 16 + col];    // A=S^T rows 16..31
    }
    acc0 = __builtin_amdgcn_wmma_f32_16x16x32_f16(false, a0, false, bf,
                                                  (short)0, acc0, false, false);
    acc1 = __builtin_amdgcn_wmma_f32_16x16x32_f16(false, a1, false, bf,
                                                  (short)0, acc1, false, false);
  }
#pragma unroll
  for (int r = 0; r < 8; ++r) {                 // C/D layout: M=r+8*half, N=col
    pLDS[r + half * 8][fcol]      = acc0[r];
    pLDS[16 + r + half * 8][fcol] = acc1[r];
  }
  __syncthreads();

  // ---- phase 2: out = LeakyReLU(pooled @ lin_w^T), K-dim = f_in (128)
  v8f o0 = {}; v8f o1 = {};
  for (int kb = 0; kb < F_ / 32; ++kb) {
    const int kf0 = kb * 32;
    v16h a0, a1, bf;
#pragma unroll
    for (int j = 0; j < 16; ++j) {
      const int kf = kf0 + kmap16(j, half);
      a0[j] = (_Float16)pLDS[col][kf];
      a1[j] = (_Float16)pLDS[16 + col][kf];
      bf[j] = (_Float16)lin_w[(size_t)fcol * F_ + kf];   // B[kf][fo]=lin_w[fo][kf]
    }
    o0 = __builtin_amdgcn_wmma_f32_16x16x32_f16(false, a0, false, bf,
                                                (short)0, o0, false, false);
    o1 = __builtin_amdgcn_wmma_f32_16x16x32_f16(false, a1, false, bf,
                                                (short)0, o1, false, false);
  }
  float* ob = out + (size_t)b * K_ * F_;
#pragma unroll
  for (int r = 0; r < 8; ++r) {
    float v0 = o0[r]; v0 = (v0 >= 0.f) ? v0 : 0.01f * v0;
    ob[(size_t)(r + half * 8) * F_ + fcol] = v0;
    float v1 = o1[r]; v1 = (v1 >= 0.f) ? v1 : 0.01f * v1;
    ob[(size_t)(16 + r + half * 8) * F_ + fcol] = v1;
  }
}

// -------------------------------------------------------------------------
extern "C" void kernel_launch(void* const* d_in, const int* in_sizes, int n_in,
                              void* d_out, int out_size, void* d_ws, size_t ws_size,
                              hipStream_t stream) {
  const float* x      = (const float*)d_in[0];   // [8,1024,128]
  const float* keys   = (const float*)d_in[1];   // [4,32,128]
  const float* conv_w = (const float*)d_in[2];   // [4]
  const float* lin_w  = (const float*)d_in[3];   // [128,128]

  float* out = (float*)d_out;                    // [8,32,128]  = 32768 floats
  float* S   = out + (size_t)B_ * K_ * F_;       // [8,1024,32] = 262144 floats

  mempool_s_kernel<<<dim3(B_ * (N_ / PTS)), dim3(128), 0, stream>>>(x, keys, conv_w, S);
  mempool_out_kernel<<<dim3(B_), dim3(256), 0, stream>>>(x, lin_w, S, out);
}
